// ControlFlexHNN_9156870275618
// MI455X (gfx1250) — compile-verified, matrix operand order checked by reading, not run on hardware
//
#include <hip/hip_runtime.h>

// ---------- types ----------
typedef __attribute__((ext_vector_type(16))) __bf16        v16bf;
typedef __attribute__((ext_vector_type(8)))  float         v8f;
typedef __attribute__((ext_vector_type(4)))  unsigned int  u32x4;
typedef int v4i __attribute__((vector_size(16)));   // matches builtin's V4i

union Frag { v16bf bf; u32x4 q[2]; };

// ---------- constants ----------
#define HDIM   1024
#define TILE_M 128
#define TILE_N 128
#define TILE_K 64
#define KSTEPS (HDIM / TILE_K)

constexpr int MODE_A2 = 0;   // epilogue: +b2, tanh, s1=1-h2^2 -> s1_bf, g2hat = Wh*s1
constexpr int MODE_G1 = 1;   // epilogue: *s0 -> g1hat_bf
constexpr int MODE_F2 = 2;   // epilogue: +bf2, tanh(v)+v*s0 -> g2f_bf

constexpr int TMODE_DH  = 0; // store f32 [M,16]
constexpr int TMODE_OUT = 1; // s = v + bff + dH; J-permute/negate into d_out

// ---------- async copy to LDS (gfx1250) with safe fallback ----------
#if defined(__gfx1250__) && __has_builtin(__builtin_amdgcn_global_load_async_to_lds_b128) && __has_builtin(__builtin_amdgcn_s_wait_asynccnt)
#define USE_ASYNC 1
#else
#define USE_ASYNC 0
#endif

__device__ __forceinline__ void copy16B_to_lds(const __bf16* g, __bf16* l)
{
#if USE_ASYNC
    __builtin_amdgcn_global_load_async_to_lds_b128(
        (__attribute__((address_space(1))) v4i*)(g),
        (__attribute__((address_space(3))) v4i*)(l), 0, 0);
#else
    *(u32x4*)l = *(const u32x4*)g;
#endif
}

__device__ __forceinline__ void wait_async_copies()
{
#if USE_ASYNC
    __builtin_amdgcn_s_wait_asynccnt(0);
#endif
}

// =====================================================================
// Big bf16 GEMM: C[M,1024] = A[M,1024] x Bw[1024,1024]^T  (Bw rows = out cols)
// f32 accumulate via v_wmma_f32_16x16x32_bf16, double-buffered LDS with
// async global->LDS prefetch, fused epilogue.
// =====================================================================
template<int MODE>
__global__ __launch_bounds__(256, 2)
void gemm_hh(const __bf16* __restrict__ A,
             const __bf16* __restrict__ Bw,
             const float*  __restrict__ bias,   // b2 (A2) / bf2 (F2) / unused
             const float*  __restrict__ whv,    // Wh (A2) / unused
             const __bf16* __restrict__ s0,     // s0 (G1,F2) / unused
             __bf16*       __restrict__ out0,   // s1 (A2) / g1hat (G1) / g2f (F2)
             __bf16*       __restrict__ out1,   // g2hat (A2) / unused
             int M)
{
    (void)M;
    __shared__ __bf16 lA[2][TILE_M * TILE_K];
    __shared__ __bf16 lB[2][TILE_N * TILE_K];

    const int tid   = threadIdx.x;
    const int lane  = tid & 31;          // wave32
    const int wid   = tid >> 5;          // 8 waves
    const int wm    = wid & 1;           // 2 row-groups of 64
    const int wn    = wid >> 1;          // 4 col-groups of 32
    const int half  = lane >> 4;
    const int l15   = lane & 15;
    const int tileM = blockIdx.x * TILE_M;
    const int tileN = blockIdx.y * TILE_N;

    v8f acc[4][2] = {};

    // copy mapping: each thread moves 32 contiguous bf16 (4 x 16B) per tile
    const int pbase = tid * 32;          // 0..8160
    const int prow  = pbase >> 6;        // /TILE_K
    const int pcol  = pbase & (TILE_K - 1);

    const __bf16* gArow = &A [(size_t)(tileM + prow) * HDIM + pcol];
    const __bf16* gBrow = &Bw[(size_t)(tileN + prow) * HDIM + pcol];

    auto prefetch = [&](int buf, int k0) {
        const __bf16* gA = gArow + k0;
        const __bf16* gB = gBrow + k0;
        __bf16* dA = &lA[buf][prow * TILE_K + pcol];
        __bf16* dB = &lB[buf][prow * TILE_K + pcol];
        #pragma unroll
        for (int c = 0; c < 4; ++c) {
            copy16B_to_lds(gA + c * 8, dA + c * 8);
            copy16B_to_lds(gB + c * 8, dB + c * 8);
        }
    };

    prefetch(0, 0);

    for (int it = 0; it < KSTEPS; ++it) {
        const int buf = it & 1;
        wait_async_copies();          // my async fills of lA/lB[buf] complete
        __syncthreads();              // everyone's fills done; prev compute retired
        if (it + 1 < KSTEPS) prefetch(buf ^ 1, (it + 1) * TILE_K);

        #pragma unroll
        for (int kk = 0; kk < TILE_K; kk += 32) {
            Frag afr[4], bfr[2];
            #pragma unroll
            for (int mi = 0; mi < 4; ++mi) {
                int r = wm * 64 + mi * 16 + l15;
                afr[mi].q[0] = *(const u32x4*)&lA[buf][r * TILE_K + kk +      8 * half];
                afr[mi].q[1] = *(const u32x4*)&lA[buf][r * TILE_K + kk + 16 + 8 * half];
            }
            #pragma unroll
            for (int ni = 0; ni < 2; ++ni) {
                int r = wn * 32 + ni * 16 + l15;
                bfr[ni].q[0] = *(const u32x4*)&lB[buf][r * TILE_K + kk +      8 * half];
                bfr[ni].q[1] = *(const u32x4*)&lB[buf][r * TILE_K + kk + 16 + 8 * half];
            }
            #pragma unroll
            for (int mi = 0; mi < 4; ++mi)
                #pragma unroll
                for (int ni = 0; ni < 2; ++ni)
                    acc[mi][ni] = __builtin_amdgcn_wmma_f32_16x16x32_bf16(
                        false, afr[mi].bf, false, bfr[ni].bf,
                        (short)0, acc[mi][ni], false, false);
        }
    }

    #pragma unroll
    for (int mi = 0; mi < 4; ++mi) {
        #pragma unroll
        for (int ni = 0; ni < 2; ++ni) {
            #pragma unroll
            for (int r = 0; r < 8; ++r) {
                int row = tileM + wm * 64 + mi * 16 + r + 8 * half;
                int col = tileN + wn * 32 + ni * 16 + l15;
                size_t idx = (size_t)row * HDIM + col;
                float v = acc[mi][ni][r];
                if (MODE == MODE_A2) {
                    v += bias[col];
                    float h2  = tanhf(v);
                    float s1v = 1.0f - h2 * h2;
                    out0[idx] = (__bf16)s1v;
                    out1[idx] = (__bf16)(whv[col] * s1v);
                } else if (MODE == MODE_G1) {
                    out0[idx] = (__bf16)(v * (float)s0[idx]);
                } else { // MODE_F2
                    v += bias[col];
                    out0[idx] = (__bf16)(tanhf(v) + v * (float)s0[idx]);
                }
            }
        }
    }
}

// =====================================================================
// Thin GEMM: C[M,16] = A[M,1024] x B16[16,1024]^T ; one WMMA tile per wave.
// =====================================================================
template<int MODE>
__global__ __launch_bounds__(256)
void gemm_thin(const __bf16* __restrict__ A,
               const __bf16* __restrict__ B16,   // [16,1024]
               const float*  __restrict__ dH,    // (OUT) [M,16]
               const float*  __restrict__ bff,   // (OUT) [16]
               float*        __restrict__ out,   // DH: dH buffer ; OUT: d_out
               int M)
{
    (void)M;
    const int tid   = threadIdx.x;
    const int lane  = tid & 31;
    const int wid   = tid >> 5;
    const int half  = lane >> 4;
    const int l15   = lane & 15;
    const int tileM = blockIdx.x * 128 + wid * 16;

    v8f acc = {};
    const __bf16* aRow = &A  [(size_t)(tileM + l15) * HDIM];
    const __bf16* bRow = &B16[(size_t)l15 * HDIM];

    for (int k0 = 0; k0 < HDIM; k0 += 32) {
        Frag a, b;
        a.q[0] = *(const u32x4*)&aRow[k0 +      8 * half];
        a.q[1] = *(const u32x4*)&aRow[k0 + 16 + 8 * half];
        b.q[0] = *(const u32x4*)&bRow[k0 +      8 * half];
        b.q[1] = *(const u32x4*)&bRow[k0 + 16 + 8 * half];
        acc = __builtin_amdgcn_wmma_f32_16x16x32_bf16(
            false, a.bf, false, b.bf, (short)0, acc, false, false);
    }

    #pragma unroll
    for (int r = 0; r < 8; ++r) {
        int row = tileM + r + 8 * half;
        int col = l15;
        float v = acc[r];
        if (MODE == TMODE_DH) {
            out[(size_t)row * 16 + col] = v;
        } else {
            float s = v + bff[col] + dH[(size_t)row * 16 + col];
            // dz_dt = [s_bot ; -s_top]
            out[(size_t)row * 16 + ((col + 8) & 15)] = (col < 8) ? -s : s;
        }
    }
}

// =====================================================================
// Small helper kernels (weight convert/transpose + small-K stages)
// =====================================================================
__global__ void k_convert(const float* __restrict__ src, __bf16* __restrict__ dst, int n)
{
    int i = blockIdx.x * blockDim.x + threadIdx.x;
    if (i < n) dst[i] = (__bf16)src[i];
}

__global__ void k_transpose_bf(const float* __restrict__ src, __bf16* __restrict__ dst,
                               int R, int C)
{
    int i = blockIdx.x * blockDim.x + threadIdx.x;
    if (i < R * C) {
        int r = i / C, c = i % C;
        dst[(size_t)c * R + r] = (__bf16)src[i];
    }
}

__global__ void k_policy_u(const float* __restrict__ z, const float* __restrict__ Wp,
                           const float* __restrict__ bp, float* __restrict__ u, int N)
{
    int n = blockIdx.x * blockDim.x + threadIdx.x;
    if (n >= N) return;
    float zr[16];
    #pragma unroll
    for (int k = 0; k < 16; ++k) zr[k] = z[(size_t)n * 16 + k];
    #pragma unroll
    for (int a = 0; a < 4; ++a) {
        float s = bp[a];
        #pragma unroll
        for (int k = 0; k < 16; ++k) s += zr[k] * Wp[a * 16 + k];
        u[(size_t)n * 4 + a] = tanhf(s);
    }
}

__global__ void k_h1(const float* __restrict__ z, const float* __restrict__ W1,
                     const float* __restrict__ b1,
                     __bf16* __restrict__ h1, __bf16* __restrict__ s0, int total)
{
    int i = blockIdx.x * blockDim.x + threadIdx.x;
    if (i >= total) return;
    int n = i >> 10, j = i & (HDIM - 1);
    const float* zr = &z[(size_t)n * 16];
    const float* wr = &W1[(size_t)j * 16];
    float s = b1[j];
    #pragma unroll
    for (int k = 0; k < 16; ++k) s += zr[k] * wr[k];
    float h = tanhf(s);
    h1[i] = (__bf16)h;
    s0[i] = (__bf16)(1.0f - h * h);
}

__global__ void k_f1(const float* __restrict__ z, const float* __restrict__ u,
                     const float* __restrict__ Wf1, const float* __restrict__ bf1,
                     const __bf16* __restrict__ s1, __bf16* __restrict__ g1f, int total)
{
    int i = blockIdx.x * blockDim.x + threadIdx.x;
    if (i >= total) return;
    int n = i >> 10, j = i & (HDIM - 1);
    const float* wr = &Wf1[(size_t)j * 20];
    const float* zr = &z[(size_t)n * 16];
    float s = bf1[j];
    #pragma unroll
    for (int k = 0; k < 16; ++k) s += zr[k] * wr[k];
    const float* ur = &u[(size_t)n * 4];
    #pragma unroll
    for (int a = 0; a < 4; ++a) s += ur[a] * wr[16 + a];
    float f = s;
    g1f[i] = (__bf16)(tanhf(f) + f * (float)s1[i]);
}

// =====================================================================
// Launch
// =====================================================================
extern "C" void kernel_launch(void* const* d_in, const int* in_sizes, int n_in,
                              void* d_out, int out_size, void* d_ws, size_t ws_size,
                              hipStream_t stream)
{
    (void)n_in; (void)out_size; (void)ws_size;

    const float* z   = (const float*)d_in[1];
    const float* W1  = (const float*)d_in[2];
    const float* b1  = (const float*)d_in[3];
    const float* W2  = (const float*)d_in[4];
    const float* b2  = (const float*)d_in[5];
    const float* Wh  = (const float*)d_in[6];
    const float* Wf1 = (const float*)d_in[8];
    const float* bf1 = (const float*)d_in[9];
    const float* Wf2 = (const float*)d_in[10];
    const float* bf2 = (const float*)d_in[11];
    const float* Wff = (const float*)d_in[12];
    const float* bff = (const float*)d_in[13];
    const float* Wp  = (const float*)d_in[14];
    const float* bp  = (const float*)d_in[15];
    float* out = (float*)d_out;

    const int N = in_sizes[1] / 16;   // 32768
    const int H = HDIM;

    // workspace carve-out (256B aligned)
    char* ws = (char*)d_ws;
    size_t off = 0;
    auto alloc = [&](size_t bytes) -> void* {
        void* p = ws + off;
        off = (off + bytes + 255) & ~(size_t)255;
        return p;
    };
    __bf16* W2bf  = (__bf16*)alloc((size_t)H * H * 2);
    __bf16* W2Tbf = (__bf16*)alloc((size_t)H * H * 2);
    __bf16* Wf2bf = (__bf16*)alloc((size_t)H * H * 2);
    __bf16* W1Tbf = (__bf16*)alloc((size_t)16 * H * 2);
    __bf16* Wffbf = (__bf16*)alloc((size_t)16 * H * 2);
    float*  u     = (float*) alloc((size_t)N * 4 * 4);
    float*  dH    = (float*) alloc((size_t)N * 16 * 4);
    __bf16* h1    = (__bf16*)alloc((size_t)N * H * 2);  // reused as g1f
    __bf16* s0    = (__bf16*)alloc((size_t)N * H * 2);
    __bf16* s1    = (__bf16*)alloc((size_t)N * H * 2);
    __bf16* g2h   = (__bf16*)alloc((size_t)N * H * 2);  // reused as g2f
    __bf16* g1h   = (__bf16*)alloc((size_t)N * H * 2);

    const int thr = 256;
    // weight prep (bf16 + required transposes)
    k_convert<<<(H * H + thr - 1) / thr, thr, 0, stream>>>(W2,  W2bf,  H * H);
    k_convert<<<(H * H + thr - 1) / thr, thr, 0, stream>>>(Wf2, Wf2bf, H * H);
    k_convert<<<(16 * H + thr - 1) / thr, thr, 0, stream>>>(Wff, Wffbf, 16 * H);
    k_transpose_bf<<<(H * H + thr - 1) / thr, thr, 0, stream>>>(W2, W2Tbf, H, H);
    k_transpose_bf<<<(H * 16 + thr - 1) / thr, thr, 0, stream>>>(W1, W1Tbf, H, 16);

    // small-K input stages
    k_policy_u<<<(N + thr - 1) / thr, thr, 0, stream>>>(z, Wp, bp, u, N);
    k_h1<<<((size_t)N * H + thr - 1) / thr, thr, 0, stream>>>(z, W1, b1, h1, s0, N * H);

    dim3 gBig(N / TILE_M, H / TILE_N);
    // a2 = h1 @ W2^T + b2 -> s1, g2hat = Wh*s1
    gemm_hh<MODE_A2><<<gBig, 256, 0, stream>>>(h1, W2bf, b2, Wh, nullptr, s1, g2h, N);
    // g1hat = (g2hat @ W2) * s0
    gemm_hh<MODE_G1><<<gBig, 256, 0, stream>>>(g2h, W2Tbf, nullptr, nullptr, s0, g1h, nullptr, N);
    // dH = g1hat @ W1
    gemm_thin<TMODE_DH><<<N / 128, 256, 0, stream>>>(g1h, W1Tbf, nullptr, nullptr, dH, N);
    // g1f = tanh(f1) + f1*s1   (f1 = [z,u] @ Wf1^T + bf1), reuse h1 buffer
    k_f1<<<((size_t)N * H + thr - 1) / thr, thr, 0, stream>>>(z, u, Wf1, bf1, s1, h1, N * H);
    // g2f = tanh(f2) + f2*s0   (f2 = g1f @ Wf2^T + bf2), reuse g2h buffer
    gemm_hh<MODE_F2><<<gBig, 256, 0, stream>>>(h1, Wf2bf, bf2, nullptr, s0, g2h, nullptr, N);
    // out = J(g2f @ Wff^T + bff + dH)
    gemm_thin<TMODE_OUT><<<N / 128, 256, 0, stream>>>(g2h, Wffbf, dH, bff, out, N);
}
